// KimiLinearSelfAttention_23055384445803
// MI455X (gfx1250) — matrix operand order repaired and checked by reading, not compiled, not run on hardware
//
#include <hip/hip_runtime.h>
#include <hip/hip_bf16.h>

// ---------------------------------------------------------------------------
// Kimi linear attention (KDA) for gfx1250 / MI455X.
// GEMMs run on the WMMA pipe (bf16 x bf16 -> f32, v_wmma_f32_16x16x32_bf16),
// 4 accumulators per wave (16x64 strip) for A-fragment reuse + pipe overlap.
// Recurrence runs per-head with the 128x128 state in LDS.
// ---------------------------------------------------------------------------

typedef __attribute__((ext_vector_type(16))) __bf16        v16bf;
typedef __attribute__((ext_vector_type(16))) unsigned short v16u;
typedef __attribute__((ext_vector_type(8)))  float          v8f;

#define S_LEN 2048
#define D_DIM 1024
#define H_N   8
#define DK_N  128
#define DV_N  128

// ---------------------------------------------------------------------------
// fp32 -> bf16 (round to nearest even), stored as raw u16
// ---------------------------------------------------------------------------
__global__ void cast_f32_bf16(const float* __restrict__ in,
                              unsigned short* __restrict__ out, size_t n) {
  size_t i = (size_t)blockIdx.x * blockDim.x + threadIdx.x;
  size_t stride = (size_t)gridDim.x * blockDim.x;
  for (; i < n; i += stride) {
    unsigned u = __builtin_bit_cast(unsigned, in[i]);
    unsigned r = (u + 0x7FFFu + ((u >> 16) & 1u)) >> 16;
    out[i] = (unsigned short)r;
  }
}

// ---------------------------------------------------------------------------
// WMMA bf16 GEMM:  C[M,N](f32) = A[M,K](bf16,row-major) @ W[N,K](bf16)^T
// Each wave computes a 16x64 strip: 4 independent accumulators sharing one
// A fragment (register-level reuse, 4 parallel WMMA chains).
// Per-lane layouts (A/B in this code's contiguous K order k = 16*half + e; C per CDNA5 ISA 7.12.2):
//   A 16x32 bf16 : lanes 0-15 row M=lane, K half 0; lanes 16-31 same rows, K half 1
//   B 32x16 bf16 : lane n = column N, lanes 0-15 hold K=0..15, lanes 16-31 K=16..31
//   C 16x16 f32  : VGPR r -> M = r + 8*(lane>=16), N = lane&15
// ---------------------------------------------------------------------------
__global__ void gemm_bf16_wmma(const unsigned short* __restrict__ A,
                               const unsigned short* __restrict__ W,
                               float* __restrict__ C,
                               int M, int N, int K) {
  const int lane   = threadIdx.x & 31;
  const int wave   = threadIdx.x >> 5;
  const int groupsN = N >> 6;                       // 64-wide column groups
  const int group = blockIdx.x * (blockDim.x >> 5) + wave;
  if (group >= (M >> 4) * groupsN) return;
  const int tm = (group / groupsN) << 4;
  const int tn = (group % groupsN) << 6;

  const int m    = lane & 15;     // A row within tile / B column within tile
  const int half = lane >> 4;     // K-half selector

  const unsigned short* arow  = A + (size_t)(tm + m) * K;        // A row (K-major)
  const unsigned short* wcol0 = W + (size_t)(tn + m) * K;        // B columns = W rows
  const unsigned short* wcol1 = wcol0 + (size_t)16 * K;
  const unsigned short* wcol2 = wcol0 + (size_t)32 * K;
  const unsigned short* wcol3 = wcol0 + (size_t)48 * K;

  v8f acc0 = {}, acc1 = {}, acc2 = {}, acc3 = {};
  for (int kk = 0; kk < K; kk += 32) {
    if (kk + 128 < K) {
      __builtin_prefetch(arow  + kk + 128, 0, 1);   // global_prefetch_b8
      __builtin_prefetch(wcol0 + kk + 128, 0, 1);
    }
    v16u av, bv0, bv1, bv2, bv3;
    // A operand: elements 0..15 -> K = kbase+0..15 (kbase = kk + 16*half)
    const int kbase = kk + half * 16;
#pragma unroll
    for (int e = 0; e < 8; ++e) {
      av[e]     = arow[kbase + e];
      av[e + 8] = arow[kbase + 8 + e];
    }
    // B operands: lanes 0-15 cover K = kk+0..15, lanes 16-31 cover K = kk+16..31
    const int kb2 = kk + half * 16;
#pragma unroll
    for (int e = 0; e < 16; ++e) {
      bv0[e] = wcol0[kb2 + e];
      bv1[e] = wcol1[kb2 + e];
      bv2[e] = wcol2[kb2 + e];
      bv3[e] = wcol3[kb2 + e];
    }
    const v16bf a = __builtin_bit_cast(v16bf, av);
    acc0 = __builtin_amdgcn_wmma_f32_16x16x32_bf16(
        false, a, false, __builtin_bit_cast(v16bf, bv0), (short)0, acc0, false, false);
    acc1 = __builtin_amdgcn_wmma_f32_16x16x32_bf16(
        false, a, false, __builtin_bit_cast(v16bf, bv1), (short)0, acc1, false, false);
    acc2 = __builtin_amdgcn_wmma_f32_16x16x32_bf16(
        false, a, false, __builtin_bit_cast(v16bf, bv2), (short)0, acc2, false, false);
    acc3 = __builtin_amdgcn_wmma_f32_16x16x32_bf16(
        false, a, false, __builtin_bit_cast(v16bf, bv3), (short)0, acc3, false, false);
  }
  const int row0 = tm + 8 * half;
  const int col  = lane & 15;
#pragma unroll
  for (int r = 0; r < 8; ++r) {
    float* crow = C + (size_t)(row0 + r) * N + tn + col;
    crow[0]  = acc0[r];
    crow[16] = acc1[r];
    crow[32] = acc2[r];
    crow[48] = acc3[r];
  }
}

// ---------------------------------------------------------------------------
// Depthwise causal conv1d (K=4) + SiLU.   y[s,c] = silu(sum_i x[s-3+i,c]*w[c,i])
// ---------------------------------------------------------------------------
__global__ void conv_silu_kernel(const float* __restrict__ xin,
                                 const float* __restrict__ wconv,
                                 float* __restrict__ y, int Sv, int Cv) {
  size_t n = (size_t)Sv * Cv;
  size_t i = (size_t)blockIdx.x * blockDim.x + threadIdx.x;
  size_t stride = (size_t)gridDim.x * blockDim.x;
  for (; i < n; i += stride) {
    int s = (int)(i / Cv), c = (int)(i % Cv);
    float acc = 0.f;
#pragma unroll
    for (int t = 0; t < 4; ++t) {
      int sp = s - 3 + t;
      if (sp >= 0) acc += xin[(size_t)sp * Cv + c] * wconv[c * 4 + t];
    }
    y[i] = acc / (1.f + __expf(-acc));
  }
}

// v = lam0 * v + lam1 * ve_v   (in place on v)
__global__ void mix_v_kernel(float* __restrict__ v, const float* __restrict__ vev,
                             const float* __restrict__ lam, size_t n) {
  size_t i = (size_t)blockIdx.x * blockDim.x + threadIdx.x;
  size_t stride = (size_t)gridDim.x * blockDim.x;
  float l0 = lam[0], l1 = lam[1];
  for (; i < n; i += stride) v[i] = l0 * v[i] + l1 * vev[i];
}

// g = -exp(A_log[h]) * softplus(glin + dt_bias)   (in place)
__global__ void gate_g_kernel(float* __restrict__ g,
                              const float* __restrict__ A_log,
                              const float* __restrict__ dt_bias, size_t n) {
  size_t i = (size_t)blockIdx.x * blockDim.x + threadIdx.x;
  size_t stride = (size_t)gridDim.x * blockDim.x;
  for (; i < n; i += stride) {
    int c = (int)(i & (D_DIM - 1));
    int h = c >> 7;
    float t = g[i] + dt_bias[c];
    float sp = (t > 20.f) ? t : logf(1.f + __expf(t));
    g[i] = -__expf(A_log[h]) * sp;
  }
}

// beta[s,h] = sigmoid(x[s,:] . Wb[h,:])
__global__ void beta_kernel(const float* __restrict__ x,
                            const float* __restrict__ Wb,
                            float* __restrict__ beta, int Sv) {
  int idx = blockIdx.x * blockDim.x + threadIdx.x;
  if (idx >= Sv * H_N) return;
  int s = idx >> 3, h = idx & 7;
  const float* xr = x + (size_t)s * D_DIM;
  const float* wr = Wb + (size_t)h * D_DIM;
  float acc = 0.f;
  for (int d = 0; d < D_DIM; ++d) acc += xr[d] * wr[d];
  beta[idx] = 1.f / (1.f + __expf(-acc));
}

// ---------------------------------------------------------------------------
// KDA recurrence, one workgroup per head, 256 threads (8 waves, wave32).
// State S[128,128] kept in LDS with a 129-stride pad (64 banks, f32).
// Per step: L2-norm q/k, decay S by exp(g), kS reduction, delta update, qS.
// ---------------------------------------------------------------------------
#define SROW 129
__global__ void kda_kernel(const float* __restrict__ q, const float* __restrict__ k,
                           const float* __restrict__ v, const float* __restrict__ g,
                           const float* __restrict__ beta, float* __restrict__ o,
                           int Sv) {
  extern __shared__ float lds[];
  float* Sst  = lds;                    // 128 * 129
  float* qs   = Sst + 128 * SROW;       // 128
  float* ks   = qs + 128;               // 128
  float* vs   = ks + 128;               // 128
  float* egs  = vs + 128;               // 128
  float* vnew = egs + 128;              // 128
  float* red  = vnew + 128;             // 256

  const int t = threadIdx.x;            // 0..255
  const int h = blockIdx.x;             // head

  for (int i = t; i < 128 * SROW; i += 256) Sst[i] = 0.f;
  __syncthreads();

  const int vcol = t >> 1;              // 0..127 : output channel for reductions
  const int halfk = t & 1;              // which 64-wide K slab
  const int kb = halfk * 64;

  for (int s = 0; s < Sv; ++s) {
    const size_t base = ((size_t)s * H_N + h) * 128;
    const float bscale = beta[(size_t)s * H_N + h];

    // --- stage raw vectors + squared norms ---
    if (t < 128) {
      float qr = q[base + t], kr = k[base + t];
      qs[t] = qr; ks[t] = kr;
      vs[t] = v[base + t];
      egs[t] = __expf(g[base + t]);
      red[t] = qr * qr;
      red[128 + t] = kr * kr;
    }
    __syncthreads();
    for (int off = 64; off > 0; off >>= 1) {
      if (t < off) red[t] += red[t + off];
      if (t >= 128 && t < 128 + off) red[t] += red[t + off];
      __syncthreads();
    }
    const float qn = rsqrtf(red[0] + 1e-6f) * 0.08838834764831845f; // *DK^-0.5
    const float kn = rsqrtf(red[128] + 1e-6f);
    __syncthreads();
    if (t < 128) { qs[t] *= qn; ks[t] *= kn; }
    __syncthreads();

    // --- decay: S[k,v] *= exp(g[k]) ; each thread owns 64 strided elements ---
#pragma unroll 4
    for (int j = 0; j < 64; ++j) {
      int idx = t + 256 * j;
      Sst[(idx >> 7) * SROW + (idx & 127)] *= egs[idx >> 7];
    }
    __syncthreads();

    // --- kS[v] = sum_k k[k] * S[k,v] ---
    {
      float acc = 0.f;
#pragma unroll 8
      for (int kk = 0; kk < 64; ++kk) {
        int kr = kb + kk;
        acc += ks[kr] * Sst[kr * SROW + vcol];
      }
      red[t] = acc;
    }
    __syncthreads();
    if (halfk == 0) vnew[vcol] = (vs[vcol] - (red[t] + red[t + 1])) * bscale;
    __syncthreads();

    // --- delta update: S[k,v] += k[k] * vnew[v] ---
#pragma unroll 4
    for (int j = 0; j < 64; ++j) {
      int idx = t + 256 * j;
      int kr = idx >> 7, vr = idx & 127;
      Sst[kr * SROW + vr] += ks[kr] * vnew[vr];
    }
    __syncthreads();

    // --- o[v] = sum_k q[k] * S[k,v] ---
    {
      float acc = 0.f;
#pragma unroll 8
      for (int kk = 0; kk < 64; ++kk) {
        int kr = kb + kk;
        acc += qs[kr] * Sst[kr * SROW + vcol];
      }
      red[t] = acc;
    }
    __syncthreads();
    if (halfk == 0) o[base + vcol] = red[t] + red[t + 1];
    __syncthreads();
  }
}

// ---------------------------------------------------------------------------
// FusedRMSNormGated: out = rmsnorm(o)*w * sigmoid(gate + bg2), emitted as bf16
// One 128-thread block per (s,h).
// ---------------------------------------------------------------------------
__global__ void norm_gate_kernel(const float* __restrict__ o,
                                 const float* __restrict__ gatelin,
                                 const float* __restrict__ bg2,
                                 const float* __restrict__ w_norm,
                                 unsigned short* __restrict__ outb) {
  __shared__ float red[128];
  const int sh = blockIdx.x;          // s*H + h
  const int t = threadIdx.x;          // 0..127
  const float x = o[(size_t)sh * 128 + t];
  red[t] = x * x;
  __syncthreads();
  for (int off = 64; off > 0; off >>= 1) {
    if (t < off) red[t] += red[t + off];
    __syncthreads();
  }
  const float r = rsqrtf(red[0] * (1.f / 128.f) + 1e-5f);
  const int c = ((sh & 7) << 7) + t;  // h*128 + dv
  const float gl = gatelin[(size_t)sh * 128 + t] + bg2[c];
  const float val = x * r * w_norm[t] / (1.f + __expf(-gl));
  unsigned u = __builtin_bit_cast(unsigned, val);
  outb[(size_t)sh * 128 + t] = (unsigned short)((u + 0x7FFFu + ((u >> 16) & 1u)) >> 16);
}

// ---------------------------------------------------------------------------
// Host-side launch
// ---------------------------------------------------------------------------
extern "C" void kernel_launch(void* const* d_in, const int* in_sizes, int n_in,
                              void* d_out, int out_size, void* d_ws, size_t ws_size,
                              hipStream_t stream) {
  const float* x        = (const float*)d_in[0];
  const float* ve       = (const float*)d_in[1];
  const float* sa       = (const float*)d_in[2];
  const float* Wq       = (const float*)d_in[3];
  const float* Wk       = (const float*)d_in[4];
  const float* Wv       = (const float*)d_in[5];
  const float* Wo       = (const float*)d_in[6];
  const float* wq_conv  = (const float*)d_in[7];
  const float* wk_conv  = (const float*)d_in[8];
  const float* wv_conv  = (const float*)d_in[9];
  const float* Wf1      = (const float*)d_in[10];
  const float* Wf2      = (const float*)d_in[11];
  const float* Wb       = (const float*)d_in[12];
  const float* A_log    = (const float*)d_in[13];
  const float* dt_bias  = (const float*)d_in[14];
  const float* Wg1      = (const float*)d_in[15];
  const float* Wg2      = (const float*)d_in[16];
  const float* bg2      = (const float*)d_in[17];
  const float* w_norm   = (const float*)d_in[18];
  float* out = (float*)d_out;

  const int S = S_LEN, D = D_DIM;
  const size_t SD = (size_t)S * D;
  const size_t DD = (size_t)D * D;
  const size_t DR = (size_t)128 * D;   // 128x1024 low-rank weights
  const size_t SR = (size_t)S * 128;

  // bump allocator over d_ws
  char* p = (char*)d_ws;
  auto alloc = [&](size_t bytes) -> void* {
    void* r = (void*)p;
    p += (bytes + 255) & ~(size_t)255;
    return r;
  };

  // bf16 (u16) buffers
  unsigned short* xb    = (unsigned short*)alloc(SD * 2);
  unsigned short* veb   = (unsigned short*)alloc(SD * 2);
  unsigned short* Wqb   = (unsigned short*)alloc(DD * 2);
  unsigned short* Wkb   = (unsigned short*)alloc(DD * 2);
  unsigned short* Wvb   = (unsigned short*)alloc(DD * 2);
  unsigned short* Wob   = (unsigned short*)alloc(DD * 2);
  unsigned short* Wf1b  = (unsigned short*)alloc(DR * 2);
  unsigned short* Wf2b  = (unsigned short*)alloc(DR * 2);
  unsigned short* Wg1b  = (unsigned short*)alloc(DR * 2);
  unsigned short* Wg2b  = (unsigned short*)alloc(DR * 2);
  unsigned short* fmidb = (unsigned short*)alloc(SR * 2);
  unsigned short* gmidb = (unsigned short*)alloc(SR * 2);
  unsigned short* ogb   = (unsigned short*)alloc(SD * 2);

  // f32 buffers
  float* qlin  = (float*)alloc(SD * 4);
  float* klin  = (float*)alloc(SD * 4);
  float* vlin  = (float*)alloc(SD * 4);
  float* velin = (float*)alloc(SD * 4);
  float* qc    = (float*)alloc(SD * 4);
  float* kc    = (float*)alloc(SD * 4);
  float* vc    = (float*)alloc(SD * 4);
  float* vec   = (float*)alloc(SD * 4);
  float* glin  = (float*)alloc(SD * 4);
  float* gatel = (float*)alloc(SD * 4);
  float* fmid  = (float*)alloc(SR * 4);
  float* gmid  = (float*)alloc(SR * 4);
  float* betab = (float*)alloc((size_t)S * H_N * 4);
  float* obuf  = (float*)alloc(SD * 4);

  auto cast = [&](const float* src, unsigned short* dst, size_t n) {
    int blocks = (int)((n + 255) / 256);
    if (blocks > 2048) blocks = 2048;
    cast_f32_bf16<<<blocks, 256, 0, stream>>>(src, dst, n);
  };
  auto gemm = [&](const unsigned short* A, const unsigned short* W, float* C,
                  int M, int N, int K) {
    int groups = (M >> 4) * (N >> 6);     // one 16x64 strip per wave
    gemm_bf16_wmma<<<(groups + 7) / 8, 256, 0, stream>>>(A, W, C, M, N, K);
  };

  // 1. downconvert activations + weights for the WMMA pipe
  cast(x, xb, SD);   cast(ve, veb, SD);
  cast(Wq, Wqb, DD); cast(Wk, Wkb, DD); cast(Wv, Wvb, DD); cast(Wo, Wob, DD);
  cast(Wf1, Wf1b, DR); cast(Wf2, Wf2b, DR); cast(Wg1, Wg1b, DR); cast(Wg2, Wg2b, DR);

  // 2. projections (WMMA)
  gemm(xb,  Wqb, qlin,  S, D, D);
  gemm(xb,  Wkb, klin,  S, D, D);
  gemm(xb,  Wvb, vlin,  S, D, D);
  gemm(veb, Wvb, velin, S, D, D);

  // 3. low-rank gate paths (WMMA)
  gemm(xb, Wf1b, fmid, S, 128, D);
  cast(fmid, fmidb, SR);
  gemm(fmidb, Wf2b, glin, S, D, 128);
  gemm(xb, Wg1b, gmid, S, 128, D);
  cast(gmid, gmidb, SR);
  gemm(gmidb, Wg2b, gatel, S, D, 128);

  // 4. beta
  beta_kernel<<<(S * H_N + 255) / 256, 256, 0, stream>>>(x, Wb, betab, S);

  // 5. causal conv + silu, v/ve mix
  const int ewBlocks = 2048;
  conv_silu_kernel<<<ewBlocks, 256, 0, stream>>>(qlin, wq_conv, qc, S, D);
  conv_silu_kernel<<<ewBlocks, 256, 0, stream>>>(klin, wk_conv, kc, S, D);
  conv_silu_kernel<<<ewBlocks, 256, 0, stream>>>(vlin, wv_conv, vc, S, D);
  conv_silu_kernel<<<ewBlocks, 256, 0, stream>>>(velin, wv_conv, vec, S, D);
  mix_v_kernel<<<ewBlocks, 256, 0, stream>>>(vc, vec, sa, SD);

  // 6. per-channel decay gate
  gate_g_kernel<<<ewBlocks, 256, 0, stream>>>(glin, A_log, dt_bias, SD);

  // 7. KDA recurrence: one workgroup per head, state in LDS
  size_t ldsBytes = (size_t)(128 * SROW + 5 * 128 + 256) * sizeof(float);
  kda_kernel<<<H_N, 256, ldsBytes, stream>>>(qc, kc, vc, glin, betab, obuf, S);

  // 8. gated RMS norm -> bf16
  norm_gate_kernel<<<S * H_N, 128, 0, stream>>>(obuf, gatel, bg2, w_norm, ogb);

  // 9. output projection (WMMA) straight into d_out
  gemm(ogb, Wob, out, S, D, D);

  (void)in_sizes; (void)n_in; (void)out_size; (void)ws_size;
}
